// MambaBlock_33011118637834
// MI455X (gfx1250) — matrix operand order repaired and checked by reading, not compiled, not run on hardware
//
#include <hip/hip_runtime.h>
#include <hip/hip_bf16.h>

// ---------------------------------------------------------------------------
// Mamba2 block forward for MI455X (gfx1250, wave32, WMMA).
// All matmuls run on V_WMMA_F32_16X16X32_BF16 (bf16 in, fp32 accumulate).
// ---------------------------------------------------------------------------

typedef __bf16 bf16;
typedef __attribute__((ext_vector_type(16))) __bf16 v16bf;
typedef __attribute__((ext_vector_type(8)))  __bf16 v8bf;
typedef __attribute__((ext_vector_type(8)))  float  v8f;
typedef __attribute__((ext_vector_type(4)))  unsigned int u32x4;

union FragBF { v16bf v; v8bf h[2]; };

#define BATCHN   2
#define SEQLEN   2048
#define DMODEL   1024
#define DINNER   2048
#define NHEADSN  32
#define HEADDIM  64
#define DSTATE   128
#define NCHUNK   32          // SEQLEN / CHUNK
#define CHUNKL   64
#define CONVDIM  2304        // DINNER + 2*DSTATE
#define DPROJ    4384        // 2*DINNER + 2*DSTATE + NHEADS
#define BLTOT    (BATCHN*SEQLEN)   // 4096

__device__ __forceinline__ float siluf(float x) { return x / (1.f + __expf(-x)); }

// ---------------------------------------------------------------------------
// fp32 -> bf16 conversion
// ---------------------------------------------------------------------------
__global__ void f32_to_bf16_kernel(const float* __restrict__ in, bf16* __restrict__ out, int n) {
    int i = blockIdx.x * 256 + threadIdx.x;
    if (i < n) out[i] = (bf16)in[i];
}

// ---------------------------------------------------------------------------
// Generic bf16 WMMA GEMM:  C[M][ldc] = A[M][K] * W[N][K]^T   (fp32 out)
// Block tile 128x128, K step 32, 256 threads = 8 waves, each wave 64x32 out.
// Fragment layouts per CDNA5 ISA 7.12.2:
//   A frag (16x32 bf16): lane half h holds K = h*8 + [0..8) and 16+h*8 + [0..8)
//   B frag (32x16 bf16): lane half h holds K = h*8 + [0..8) and 16+h*8 + [0..8)
//   C/D (16x16 f32): vgpr r -> row r + 8*half, col = lane&15
// ---------------------------------------------------------------------------
__global__ __launch_bounds__(256) void gemm_bf16_wmma(
    const bf16* __restrict__ A, const bf16* __restrict__ W,
    float* __restrict__ C, int M, int N, int K, int ldc)
{
    const int LDK = 40;  // padded K stride (elements); 80B rows, 16B aligned
    __shared__ __align__(16) bf16 sA[128 * 40];
    __shared__ __align__(16) bf16 sB[128 * 40];

    int bm = blockIdx.y * 128, bn = blockIdx.x * 128;
    int tid  = threadIdx.x;
    int wave = tid >> 5, lane = tid & 31;
    int hlf  = lane >> 4, l16 = lane & 15;
    int wm = wave >> 2;        // 0..1 : 64 rows
    int wn = wave & 3;         // 0..3 : 32 cols

    v8f acc[4][2];
    for (int i = 0; i < 4; i++)
        for (int j = 0; j < 2; j++)
            for (int e = 0; e < 8; e++) acc[i][j][e] = 0.f;

    int r  = tid >> 1;          // 0..127 tile row
    int kk = (tid & 1) << 4;    // 0 or 16

    for (int k0 = 0; k0 < K; k0 += 32) {
        // ---- stage A tile (rows always in range: M is a multiple of 128)
        const bf16* asrc = A + (size_t)(bm + r) * K + k0 + kk;
        ((u32x4*)&sA[r * LDK + kk])[0]     = ((const u32x4*)asrc)[0];
        ((u32x4*)&sA[r * LDK + kk + 8])[0] = ((const u32x4*)(asrc + 8))[0];
        if (k0 + 32 < K) __builtin_prefetch(asrc + 32, 0, 0);
        // ---- stage W tile (guard ragged N)
        int n = bn + r;
        if (n < N) {
            const bf16* bsrc = W + (size_t)n * K + k0 + kk;
            ((u32x4*)&sB[r * LDK + kk])[0]     = ((const u32x4*)bsrc)[0];
            ((u32x4*)&sB[r * LDK + kk + 8])[0] = ((const u32x4*)(bsrc + 8))[0];
            if (k0 + 32 < K) __builtin_prefetch(bsrc + 32, 0, 0);
        } else {
            u32x4 z = {0u, 0u, 0u, 0u};
            ((u32x4*)&sB[r * LDK + kk])[0]     = z;
            ((u32x4*)&sB[r * LDK + kk + 8])[0] = z;
        }
        __syncthreads();

        v16bf afrag[4], bfrag[2];
        for (int i = 0; i < 4; i++) {
            const bf16* p = &sA[(wm * 64 + i * 16 + l16) * LDK];
            FragBF f;
            f.h[0] = *(const v8bf*)(p + hlf * 8);
            f.h[1] = *(const v8bf*)(p + 16 + hlf * 8);
            afrag[i] = f.v;
        }
        for (int j = 0; j < 2; j++) {
            const bf16* p = &sB[(wn * 32 + j * 16 + l16) * LDK];
            FragBF f;
            f.h[0] = *(const v8bf*)(p + hlf * 8);
            f.h[1] = *(const v8bf*)(p + 16 + hlf * 8);
            bfrag[j] = f.v;
        }
        #pragma unroll
        for (int i = 0; i < 4; i++)
            #pragma unroll
            for (int j = 0; j < 2; j++)
                acc[i][j] = __builtin_amdgcn_wmma_f32_16x16x32_bf16(
                    false, afrag[i], false, bfrag[j], (short)0, acc[i][j], false, false);
        __syncthreads();
    }

    for (int j = 0; j < 2; j++) {
        int n = bn + wn * 32 + j * 16 + l16;
        if (n >= N) continue;
        for (int i = 0; i < 4; i++) {
            int mbase = bm + wm * 64 + i * 16 + hlf * 8;
            #pragma unroll
            for (int rr = 0; rr < 8; rr++)
                C[(size_t)(mbase + rr) * ldc + n] = acc[i][j][rr];
        }
    }
}

// ---------------------------------------------------------------------------
// dt = softplus(dt_raw + dt_bias);  aval = -exp(A_log) * dt     [bl][h]
// ---------------------------------------------------------------------------
__global__ void dtprep_kernel(const float* __restrict__ zx,
                              const float* __restrict__ dt_bias,
                              const float* __restrict__ A_log,
                              float* __restrict__ dtv, float* __restrict__ aval)
{
    int idx = blockIdx.x * 256 + threadIdx.x;
    if (idx >= BLTOT * NHEADSN) return;
    int bl = idx >> 5, hh = idx & 31;
    float x  = zx[(size_t)bl * DPROJ + (DINNER + CONVDIM) + hh] + dt_bias[hh];
    float sp = (x > 20.f) ? x : log1pf(__expf(x));
    dtv[idx]  = sp;
    aval[idx] = -__expf(A_log[hh]) * sp;
}

// ---------------------------------------------------------------------------
// Causal depthwise conv (width 4) + bias + SiLU, then split into bf16
// x, x*dt, B, C buffers.
// ---------------------------------------------------------------------------
__global__ void conv_silu_kernel(const float* __restrict__ zx,
                                 const float* __restrict__ conv_w,
                                 const float* __restrict__ conv_b,
                                 const float* __restrict__ dtv,
                                 bf16* __restrict__ xbf, bf16* __restrict__ xdbf,
                                 bf16* __restrict__ bbf, bf16* __restrict__ cbf)
{
    int idx = blockIdx.x * 256 + threadIdx.x;
    if (idx >= BLTOT * CONVDIM) return;
    int bl = idx / CONVDIM;
    int ch = idx - bl * CONVDIM;
    int b = bl >> 11, l = bl & 2047;
    float acc = conv_b[ch];
    #pragma unroll
    for (int j = 0; j < 4; j++) {
        int ls = l - 3 + j;
        if (ls >= 0)
            acc += zx[(size_t)(b * SEQLEN + ls) * DPROJ + DINNER + ch] * conv_w[ch * 4 + j];
    }
    float v = siluf(acc);
    if (ch < DINNER) {
        int hh = ch >> 6;
        xbf [(size_t)bl * DINNER + ch] = (bf16)v;
        xdbf[(size_t)bl * DINNER + ch] = (bf16)(v * dtv[bl * NHEADSN + hh]);
    } else if (ch < DINNER + DSTATE) {
        bbf[(size_t)bl * DSTATE + (ch - DINNER)] = (bf16)v;
    } else {
        cbf[(size_t)bl * DSTATE + (ch - DINNER - DSTATE)] = (bf16)v;
    }
}

// ---------------------------------------------------------------------------
// Per (b, chunk): G[l][s] = C[l,:] . B[s,:]   (64x64, K=128) via WMMA.
// Both operands read K-contiguous straight from global memory.
// ---------------------------------------------------------------------------
__global__ __launch_bounds__(128) void ssd_g_kernel(const bf16* __restrict__ cbf,
                                                    const bf16* __restrict__ bbf,
                                                    float* __restrict__ gbuf)
{
    int bc = blockIdx.x;
    int b = bc >> 5, c = bc & 31;
    int bl0 = b * SEQLEN + c * CHUNKL;
    int tid = threadIdx.x, wave = tid >> 5, lane = tid & 31;
    int hlf = lane >> 4, l16 = lane & 15;
    int mt = wave;
    size_t gbase = (size_t)bc * (CHUNKL * CHUNKL);
    for (int nt = 0; nt < 4; nt++) {
        v8f acc = {0.f,0.f,0.f,0.f,0.f,0.f,0.f,0.f};
        for (int ks = 0; ks < DSTATE; ks += 32) {
            FragBF fa, fb;
            const bf16* arow = &cbf[(size_t)(bl0 + mt * 16 + l16) * DSTATE + ks];
            fa.h[0] = *(const v8bf*)(arow + hlf * 8);
            fa.h[1] = *(const v8bf*)(arow + 16 + hlf * 8);
            const bf16* brow = &bbf[(size_t)(bl0 + nt * 16 + l16) * DSTATE + ks];
            fb.h[0] = *(const v8bf*)(brow + hlf * 8);
            fb.h[1] = *(const v8bf*)(brow + 16 + hlf * 8);
            acc = __builtin_amdgcn_wmma_f32_16x16x32_bf16(
                false, fa.v, false, fb.v, (short)0, acc, false, false);
        }
        #pragma unroll
        for (int r = 0; r < 8; r++) {
            int l = mt * 16 + hlf * 8 + r, s = nt * 16 + l16;
            gbuf[gbase + l * CHUNKL + s] = acc[r];
        }
    }
}

// ---------------------------------------------------------------------------
// Per (b, chunk, head): in-chunk A cumsum; Y_diag = (G.*Ldec) @ Xd ;
// chunk states[p][n] = sum_l Xd[l,p] * (exp(T-Acs[l]) * B[l,n]).
// LDS staging transposes operands so all WMMA fragment loads are
// contiguous 16B ds reads.
// ---------------------------------------------------------------------------
#define GST 72  // padded LDS row stride (elements): 144B rows
__global__ __launch_bounds__(128) void ssd_chunk_kernel(
    const float* __restrict__ aval, const bf16* __restrict__ xdbf,
    const bf16* __restrict__ bbf,   const float* __restrict__ gbuf,
    float* __restrict__ ybuf, bf16* __restrict__ states,
    float* __restrict__ acsbuf, float* __restrict__ tbuf)
{
    int h = blockIdx.x & 31;
    int c = (blockIdx.x >> 5) & 31;
    int b = blockIdx.x >> 10;
    int bl0 = b * SEQLEN + c * CHUNKL;
    int bch = (b * NCHUNK + c) * NHEADSN + h;
    int tid = threadIdx.x;
    int wave = tid >> 5, lane = tid & 31, hlf = lane >> 4, l16 = lane & 15;

    __shared__ float acs[CHUNKL];
    __shared__ __align__(16) bf16 sGL[CHUNKL * GST];
    __shared__ __align__(16) bf16 sXT[CHUNKL * GST];     // Xd^T [p][l]
    __shared__ __align__(16) bf16 sBT[DSTATE * GST];     // (ds*B)^T [n][l]

    if (tid < CHUNKL) acs[tid] = aval[(size_t)(bl0 + tid) * NHEADSN + h];
    __syncthreads();
    if (tid == 0) {
        float s = 0.f;
        for (int l = 0; l < CHUNKL; l++) { s += acs[l]; acs[l] = s; }
    }
    __syncthreads();
    float T = acs[CHUNKL - 1];
    if (tid < CHUNKL) acsbuf[(size_t)bch * CHUNKL + tid] = acs[tid];
    if (tid == 0) tbuf[bch] = T;

    // stage Xd^T
    for (int idx = tid; idx < CHUNKL * HEADDIM; idx += 128) {
        int l = idx >> 6, p = idx & 63;
        sXT[p * GST + l] = xdbf[(size_t)(bl0 + l) * DINNER + h * HEADDIM + p];
    }
    // stage (decay_states * B)^T
    for (int idx = tid; idx < CHUNKL * DSTATE; idx += 128) {
        int l = idx >> 7, n = idx & 127;
        float ds = __expf(T - acs[l]);
        sBT[n * GST + l] = (bf16)((float)bbf[(size_t)(bl0 + l) * DSTATE + n] * ds);
    }
    // stage masked/decayed G
    {
        size_t gbase = (size_t)(b * NCHUNK + c) * (CHUNKL * CHUNKL);
        for (int idx = tid; idx < CHUNKL * CHUNKL; idx += 128) {
            int l = idx >> 6, s = idx & 63;
            float g = (l >= s) ? gbuf[gbase + idx] * __expf(acs[l] - acs[s]) : 0.f;
            sGL[l * GST + s] = (bf16)g;
        }
    }
    __syncthreads();

    int mt = wave;
    // ---- Y_diag = GL @ Xd : M=l(64), N=p(64), K=s(64)
    for (int nt = 0; nt < 4; nt++) {
        v8f acc = {0.f,0.f,0.f,0.f,0.f,0.f,0.f,0.f};
        for (int ks = 0; ks < CHUNKL; ks += 32) {
            FragBF fa, fb;
            const bf16* arow = &sGL[(mt * 16 + l16) * GST + ks];
            fa.h[0] = *(const v8bf*)(arow + hlf * 8);
            fa.h[1] = *(const v8bf*)(arow + 16 + hlf * 8);
            const bf16* brow = &sXT[(nt * 16 + l16) * GST + ks];
            fb.h[0] = *(const v8bf*)(brow + hlf * 8);
            fb.h[1] = *(const v8bf*)(brow + 16 + hlf * 8);
            acc = __builtin_amdgcn_wmma_f32_16x16x32_bf16(
                false, fa.v, false, fb.v, (short)0, acc, false, false);
        }
        #pragma unroll
        for (int r = 0; r < 8; r++) {
            int l = mt * 16 + hlf * 8 + r, p = nt * 16 + l16;
            ybuf[(size_t)(bl0 + l) * DINNER + h * HEADDIM + p] = acc[r];
        }
    }
    // ---- states[p][n] = Xd^T @ (ds*B) : M=p(64), N=n(128), K=l(64)
    size_t sbase = (size_t)bch * (HEADDIM * DSTATE);
    for (int nt = 0; nt < 8; nt++) {
        v8f acc = {0.f,0.f,0.f,0.f,0.f,0.f,0.f,0.f};
        for (int ks = 0; ks < CHUNKL; ks += 32) {
            FragBF fa, fb;
            const bf16* arow = &sXT[(mt * 16 + l16) * GST + ks];
            fa.h[0] = *(const v8bf*)(arow + hlf * 8);
            fa.h[1] = *(const v8bf*)(arow + 16 + hlf * 8);
            const bf16* brow = &sBT[(nt * 16 + l16) * GST + ks];
            fb.h[0] = *(const v8bf*)(brow + hlf * 8);
            fb.h[1] = *(const v8bf*)(brow + 16 + hlf * 8);
            acc = __builtin_amdgcn_wmma_f32_16x16x32_bf16(
                false, fa.v, false, fb.v, (short)0, acc, false, false);
        }
        #pragma unroll
        for (int r = 0; r < 8; r++) {
            int p = mt * 16 + hlf * 8 + r, n = nt * 16 + l16;
            states[sbase + p * DSTATE + n] = (bf16)acc[r];
        }
    }
}

// ---------------------------------------------------------------------------
// Inter-chunk state recurrence per (b,h):  S_{c+1} = exp(T_c)*S_c + states_c.
// Writes the carried state entering each chunk (bf16, [p][n]).
// ---------------------------------------------------------------------------
__global__ __launch_bounds__(256) void ssd_recur_kernel(const bf16* __restrict__ states,
                                                        const float* __restrict__ tbuf,
                                                        bf16* __restrict__ sout)
{
    int b = blockIdx.x >> 5;
    int h = blockIdx.x & 31;
    int t = threadIdx.x;
    float S[32];
    #pragma unroll
    for (int i = 0; i < 32; i++) S[i] = 0.f;
    for (int c = 0; c < NCHUNK; c++) {
        size_t base = (size_t)((b * NCHUNK + c) * NHEADSN + h) * (HEADDIM * DSTATE);
        #pragma unroll
        for (int i = 0; i < 32; i++) sout[base + t + i * 256] = (bf16)S[i];
        float dec = __expf(tbuf[(b * NCHUNK + c) * NHEADSN + h]);
        #pragma unroll
        for (int i = 0; i < 32; i++)
            S[i] = S[i] * dec + (float)states[base + t + i * 256];
    }
}

// ---------------------------------------------------------------------------
// Y_off[l][p] = exp(Acs[l]) * (C @ S^T)[l][p], accumulated into ybuf.
// M=l(64), N=p(64), K=n(128). S stored [p][n] -> B fragment contiguous.
// ---------------------------------------------------------------------------
__global__ __launch_bounds__(128) void ssd_yoff_kernel(const bf16* __restrict__ cbf,
                                                       const bf16* __restrict__ sbf,
                                                       const float* __restrict__ acsbuf,
                                                       float* __restrict__ ybuf)
{
    int h = blockIdx.x & 31;
    int c = (blockIdx.x >> 5) & 31;
    int b = blockIdx.x >> 10;
    int bl0 = b * SEQLEN + c * CHUNKL;
    int bch = (b * NCHUNK + c) * NHEADSN + h;
    int tid = threadIdx.x;
    __shared__ float acs[CHUNKL];
    if (tid < CHUNKL) acs[tid] = acsbuf[(size_t)bch * CHUNKL + tid];
    __syncthreads();
    int wave = tid >> 5, lane = tid & 31, hlf = lane >> 4, l16 = lane & 15;
    int mt = wave;
    size_t sbase = (size_t)bch * (HEADDIM * DSTATE);
    for (int nt = 0; nt < 4; nt++) {
        v8f acc = {0.f,0.f,0.f,0.f,0.f,0.f,0.f,0.f};
        for (int ks = 0; ks < DSTATE; ks += 32) {
            FragBF fa, fb;
            const bf16* arow = &cbf[(size_t)(bl0 + mt * 16 + l16) * DSTATE + ks];
            fa.h[0] = *(const v8bf*)(arow + hlf * 8);
            fa.h[1] = *(const v8bf*)(arow + 16 + hlf * 8);
            const bf16* brow = &sbf[sbase + (size_t)(nt * 16 + l16) * DSTATE + ks];
            fb.h[0] = *(const v8bf*)(brow + hlf * 8);
            fb.h[1] = *(const v8bf*)(brow + 16 + hlf * 8);
            acc = __builtin_amdgcn_wmma_f32_16x16x32_bf16(
                false, fa.v, false, fb.v, (short)0, acc, false, false);
        }
        #pragma unroll
        for (int r = 0; r < 8; r++) {
            int l = mt * 16 + hlf * 8 + r, p = nt * 16 + l16;
            size_t oi = (size_t)(bl0 + l) * DINNER + h * HEADDIM + p;
            ybuf[oi] += acc[r] * __expf(acs[l]);
        }
    }
}

// ---------------------------------------------------------------------------
// y = (Y + D*x) * silu(z), RMSNorm over DINNER, write bf16 for out_proj.
// ---------------------------------------------------------------------------
__global__ __launch_bounds__(256) void gate_norm_kernel(const float* __restrict__ Ybuf,
                                                        const bf16* __restrict__ xbf,
                                                        const float* __restrict__ zx,
                                                        const float* __restrict__ Dp,
                                                        const float* __restrict__ norm_w,
                                                        bf16* __restrict__ ybf)
{
    int bl = blockIdx.x;
    int t  = threadIdx.x;
    __shared__ float red[256];
    float yv[8];
    float ss = 0.f;
    #pragma unroll
    for (int i = 0; i < 8; i++) {
        int ch = t + i * 256;
        int hh = ch >> 6;
        float y = Ybuf[(size_t)bl * DINNER + ch] + Dp[hh] * (float)xbf[(size_t)bl * DINNER + ch];
        float z = zx[(size_t)bl * DPROJ + ch];
        y *= siluf(z);
        yv[i] = y;
        ss += y * y;
    }
    red[t] = ss;
    __syncthreads();
    for (int s = 128; s > 0; s >>= 1) {
        if (t < s) red[t] += red[t + s];
        __syncthreads();
    }
    float scale = rsqrtf(red[0] / (float)DINNER + 1e-5f);
    #pragma unroll
    for (int i = 0; i < 8; i++) {
        int ch = t + i * 256;
        ybf[(size_t)bl * DINNER + ch] = (bf16)(yv[i] * scale * norm_w[ch]);
    }
}

// ---------------------------------------------------------------------------
extern "C" void kernel_launch(void* const* d_in, const int* in_sizes, int n_in,
                              void* d_out, int out_size, void* d_ws, size_t ws_size,
                              hipStream_t stream)
{
    const float* u        = (const float*)d_in[0];
    const float* in_w     = (const float*)d_in[1];
    const float* conv_w   = (const float*)d_in[2];
    const float* conv_b   = (const float*)d_in[3];
    const float* dt_bias  = (const float*)d_in[4];
    const float* A_log    = (const float*)d_in[5];
    const float* Dp       = (const float*)d_in[6];
    const float* norm_w   = (const float*)d_in[7];
    const float* out_w    = (const float*)d_in[8];
    float* out = (float*)d_out;

    char* ws = (char*)d_ws;
    size_t off = 0;
    auto take = [&](size_t bytes) -> char* {
        char* p = ws + off;
        off = (off + bytes + 255) & ~(size_t)255;
        return p;
    };

    bf16*  u_bf    = (bf16*) take((size_t)BLTOT * DMODEL * 2);
    bf16*  win_bf  = (bf16*) take((size_t)DPROJ * DMODEL * 2);
    bf16*  wout_bf = (bf16*) take((size_t)DMODEL * DINNER * 2);
    float* zx      = (float*)take((size_t)BLTOT * DPROJ * 4);
    float* dtv     = (float*)take((size_t)BLTOT * NHEADSN * 4);
    float* aval    = (float*)take((size_t)BLTOT * NHEADSN * 4);
    bf16*  xbf     = (bf16*) take((size_t)BLTOT * DINNER * 2);
    bf16*  xdbf    = (bf16*) take((size_t)BLTOT * DINNER * 2);
    bf16*  bbf     = (bf16*) take((size_t)BLTOT * DSTATE * 2);
    bf16*  cbf     = (bf16*) take((size_t)BLTOT * DSTATE * 2);
    float* gbuf    = (float*)take((size_t)BATCHN * NCHUNK * CHUNKL * CHUNKL * 4);
    float* ybuf    = (float*)take((size_t)BLTOT * DINNER * 4);
    bf16*  stbf    = (bf16*) take((size_t)BATCHN * NCHUNK * NHEADSN * HEADDIM * DSTATE * 2);
    bf16*  sbf     = (bf16*) take((size_t)BATCHN * NCHUNK * NHEADSN * HEADDIM * DSTATE * 2);
    float* acsbuf  = (float*)take((size_t)BATCHN * NCHUNK * NHEADSN * CHUNKL * 4);
    float* tbuf    = (float*)take((size_t)BATCHN * NCHUNK * NHEADSN * 4);
    bf16*  ybf     = (bf16*) take((size_t)BLTOT * DINNER * 2);
    (void)ws_size; (void)n_in; (void)in_sizes; (void)out_size;

    // 1. fp32 -> bf16 conversions
    {
        int n1 = BLTOT * DMODEL;
        f32_to_bf16_kernel<<<(n1 + 255) / 256, 256, 0, stream>>>(u, u_bf, n1);
        int n2 = DPROJ * DMODEL;
        f32_to_bf16_kernel<<<(n2 + 255) / 256, 256, 0, stream>>>(in_w, win_bf, n2);
        int n3 = DMODEL * DINNER;
        f32_to_bf16_kernel<<<(n3 + 255) / 256, 256, 0, stream>>>(out_w, wout_bf, n3);
    }
    // 2. in_proj GEMM: zx = u @ in_w^T  (M=4096, N=4384, K=1024)
    {
        dim3 g((DPROJ + 127) / 128, BLTOT / 128);
        gemm_bf16_wmma<<<g, 256, 0, stream>>>(u_bf, win_bf, zx, BLTOT, DPROJ, DMODEL, DPROJ);
    }
    // 3. dt / A prep
    {
        int n = BLTOT * NHEADSN;
        dtprep_kernel<<<(n + 255) / 256, 256, 0, stream>>>(zx, dt_bias, A_log, dtv, aval);
    }
    // 4. conv + silu + split
    {
        int n = BLTOT * CONVDIM;
        conv_silu_kernel<<<(n + 255) / 256, 256, 0, stream>>>(
            zx, conv_w, conv_b, dtv, xbf, xdbf, bbf, cbf);
    }
    // 5. G = C @ B^T per (b, chunk)
    ssd_g_kernel<<<BATCHN * NCHUNK, 128, 0, stream>>>(cbf, bbf, gbuf);
    // 6. per-chunk Y_diag + states
    ssd_chunk_kernel<<<BATCHN * NCHUNK * NHEADSN, 128, 0, stream>>>(
        aval, xdbf, bbf, gbuf, ybuf, stbf, acsbuf, tbuf);
    // 7. inter-chunk recurrence
    ssd_recur_kernel<<<BATCHN * NHEADSN, 256, 0, stream>>>(stbf, tbuf, sbf);
    // 8. Y_off accumulation
    ssd_yoff_kernel<<<BATCHN * NCHUNK * NHEADSN, 128, 0, stream>>>(cbf, sbf, acsbuf, ybuf);
    // 9. skip + gate + RMSNorm
    gate_norm_kernel<<<BLTOT, 256, 0, stream>>>(ybuf, xbf, zx, Dp, norm_w, ybf);
    // 10. out_proj GEMM: out = y @ out_w^T (M=4096, N=1024, K=2048)
    {
        dim3 g(DMODEL / 128, BLTOT / 128);
        gemm_bf16_wmma<<<g, 256, 0, stream>>>(ybf, wout_bf, out, BLTOT, DMODEL, DINNER, DMODEL);
    }
}
